// LinearAttention_33749853011998
// MI455X (gfx1250) — matrix operand order repaired and checked by reading, not compiled, not run on hardware
//
#include <hip/hip_runtime.h>
#include <cmath>

#define NB   8
#define NC1  384
#define NC2  192
#define NL   4096
#define NHD  8
#define NDH  48
#define NSPLIT 16

typedef __attribute__((ext_vector_type(16))) _Float16 v16h;
typedef __attribute__((ext_vector_type(8)))  _Float16 v8h;
typedef __attribute__((ext_vector_type(8)))  float    v8f;
typedef __attribute__((ext_vector_type(4)))  float    v4f;

static __device__ __forceinline__ v16h cat16(v8h a, v8h b) {
  return __builtin_shufflevector(a, b, 0,1,2,3,4,5,6,7,8,9,10,11,12,13,14,15);
}

static __device__ __forceinline__ v8f wmma_f16(v16h a, v16h b, v8f c) {
  // v_wmma_f32_16x16x32_f16: D = A(16x32) * B(32x16) + C(16x16 f32)
  return __builtin_amdgcn_wmma_f32_16x16x32_f16(false, a, false, b, (short)0, c, false, false);
}

// ---- fragment loaders (ISA 7.12.2 layouts, wave32) ----
// A 16x32 f16, row-major source, row stride ld (halfs).
// lane r=lane&15 -> row; half-group c=lane>>4 -> K chunks [8c..8c+7] and [16+8c..16+8c+7].
static __device__ __forceinline__ v16h load_a16(const _Float16* __restrict__ src, int ld) {
  int lane = threadIdx.x & 31;
  int r = lane & 15, c = lane >> 4;
  const _Float16* p = src + (long long)r * ld + c * 8;
  v8h x = *(const v8h*)p;
  v8h y = *(const v8h*)(p + 16);
  return cat16(x, y);
}
// A fragment where only K=0..15 of the 32-chunk is valid (tail zero).
static __device__ __forceinline__ v16h load_a16_tail(const _Float16* __restrict__ src, int ld) {
  int lane = threadIdx.x & 31;
  int r = lane & 15, c = lane >> 4;
  v8h x = *(const v8h*)(src + (long long)r * ld + c * 8);
  v8h z = {};
  return cat16(x, z);
}
// B 32x16 f16 loaded from B^T row-major [N x K], row stride ld.
// lane n=lane&15 -> column; half-group c -> K chunks [8c..8c+7] and [16+8c..16+8c+7].
static __device__ __forceinline__ v16h load_b16(const _Float16* __restrict__ srcT, int ld) {
  int lane = threadIdx.x & 31;
  int n = lane & 15, c = lane >> 4;
  const _Float16* p = srcT + (long long)n * ld + c * 8;
  v8h x = *(const v8h*)p;
  v8h y = *(const v8h*)(p + 16);
  return cat16(x, y);
}
// B fragment where only K=0..15 valid: elements 8..15 are zero.
static __device__ __forceinline__ v16h load_b16_tail(const _Float16* __restrict__ srcT, int ld) {
  int lane = threadIdx.x & 31;
  int n = lane & 15, c = lane >> 4;
  v16h out = {};
  {
    const _Float16* p = srcT + (long long)n * ld + c * 8;
    v8h z = {};
    out = cat16(*(const v8h*)p, z);
  }
  return out;
}

// ---------------- kernel 1: transpose + fp32->fp16 convert ----------------
// in: [rows, cols] f32 row-major (per batch). out: [cols, rows] f16.
__global__ void la_cvt_tr(const float* __restrict__ in, _Float16* __restrict__ out,
                          int rows, int cols, int nbatch) {
  long long per = (long long)rows * cols;
  long long total = per * nbatch;
  for (long long i = (long long)blockIdx.x * blockDim.x + threadIdx.x; i < total;
       i += (long long)gridDim.x * blockDim.x) {
    long long b = i / per, rem = i - b * per;
    int r = (int)(rem / cols), c = (int)(rem - (long long)r * cols);
    out[b * per + (long long)c * rows + r] = (_Float16)in[i];
  }
}

// ---------------- kernel 2: fused K/V projection (4x N-blocked) ----------------
// Each wave: one 16-token strip x 64 output channels (4 n-tiles), K and V together.
// kbuf [B, 384, L] f32 = (x2f@Wk + bk)/scale ; vbuf [B, 384, L] f16 = x2f@Wv + bv
__global__ __launch_bounds__(256) void la_proj_kv(
    const _Float16* __restrict__ x2h, const _Float16* __restrict__ WkT,
    const _Float16* __restrict__ WvT, const float* __restrict__ bk,
    const float* __restrict__ bv, float* __restrict__ kbuf,
    _Float16* __restrict__ vbuf, float inv_scale) {
  int wave = (blockIdx.x * blockDim.x + threadIdx.x) >> 5;
  int ng = wave % (NC1 / 64);            // 6 groups of 4 n-tiles
  int rem = wave / (NC1 / 64);
  int lt = rem % (NL / 16);
  int b  = rem / (NL / 16);
  const _Float16* A0 = x2h + ((long long)b * NL + lt * 16) * NC2;
  v8f ak[4] = {{}, {}, {}, {}};
  v8f av[4] = {{}, {}, {}, {}};
  for (int kk = 0; kk < NC2; kk += 32) {
    v16h a = load_a16(A0 + kk, NC2);
#pragma unroll
    for (int j = 0; j < 4; j++) {
      int nt = ng * 4 + j;
      ak[j] = wmma_f16(a, load_b16(WkT + (long long)nt * 16 * NC2 + kk, NC2), ak[j]);
      av[j] = wmma_f16(a, load_b16(WvT + (long long)nt * 16 * NC2 + kk, NC2), av[j]);
    }
  }
  int lane = threadIdx.x & 31;
  int n = lane & 15, hg = lane >> 4;
  int l0 = lt * 16 + hg * 8;             // element i -> token l0+i
#pragma unroll
  for (int j = 0; j < 4; j++) {
    int c = (ng * 4 + j) * 16 + n;
    float bkc = bk[c], bvc = bv[c];
    float* kd = kbuf + ((long long)b * NC1 + c) * NL + l0;
    _Float16* vd = vbuf + ((long long)b * NC1 + c) * NL + l0;
    v4f k0, k1; v8h vv;
    for (int i = 0; i < 4; i++) k0[i] = (ak[j][i] + bkc) * inv_scale;
    for (int i = 0; i < 4; i++) k1[i] = (ak[j][i + 4] + bkc) * inv_scale;
    for (int i = 0; i < 8; i++) vv[i] = (_Float16)(av[j][i] + bvc);
    *(v4f*)kd = k0; *(v4f*)(kd + 4) = k1;
    *(v8h*)vd = vv;
  }
}

// ---------------- kernel 3: Q projection (4x N-blocked) ----------------
// qbuf [B, L, 384] f16 = (x1f@Wq + bq)/scale  (token-major: A rows for ctx gemm)
__global__ __launch_bounds__(256) void la_proj_q(
    const _Float16* __restrict__ x1h, const _Float16* __restrict__ WqT,
    const float* __restrict__ bq, _Float16* __restrict__ qbuf, float inv_scale) {
  int wave = (blockIdx.x * blockDim.x + threadIdx.x) >> 5;
  int ng = wave % (NC1 / 64);
  int rem = wave / (NC1 / 64);
  int lt = rem % (NL / 16);
  int b  = rem / (NL / 16);
  const _Float16* A0 = x1h + ((long long)b * NL + lt * 16) * NC1;
  v8f acc[4] = {{}, {}, {}, {}};
  for (int kk = 0; kk < NC1; kk += 32) {
    v16h a = load_a16(A0 + kk, NC1);
#pragma unroll
    for (int j = 0; j < 4; j++) {
      int nt = ng * 4 + j;
      acc[j] = wmma_f16(a, load_b16(WqT + (long long)nt * 16 * NC1 + kk, NC1), acc[j]);
    }
  }
  int lane = threadIdx.x & 31;
  int n = lane & 15, hg = lane >> 4;
#pragma unroll
  for (int j = 0; j < 4; j++) {
    int c = (ng * 4 + j) * 16 + n;
    float bqc = bq[c];
    for (int i = 0; i < 8; i++) {
      int l = lt * 16 + i + 8 * hg;
      qbuf[((long long)b * NL + l) * NC1 + c] = (_Float16)((acc[j][i] + bqc) * inv_scale);
    }
  }
}

// ---------------- kernel 4: per-(b,channel) max & sum-exp over L ----------------
__global__ __launch_bounds__(256) void la_colstats(const float* __restrict__ kbuf,
                                                   float* __restrict__ mx,
                                                   float* __restrict__ sm) {
  __shared__ float red[256];
  int row = blockIdx.x;                      // b*384 + c
  const float* p = kbuf + (long long)row * NL;
  float m = -3.0e30f;
  for (int i = threadIdx.x; i < NL; i += 256) m = fmaxf(m, p[i]);
  red[threadIdx.x] = m; __syncthreads();
  for (int s = 128; s > 0; s >>= 1) {
    if (threadIdx.x < s) red[threadIdx.x] = fmaxf(red[threadIdx.x], red[threadIdx.x + s]);
    __syncthreads();
  }
  m = red[0]; __syncthreads();
  float s = 0.f;
  for (int i = threadIdx.x; i < NL; i += 256) s += __expf(p[i] - m);
  red[threadIdx.x] = s; __syncthreads();
  for (int st = 128; st > 0; st >>= 1) {
    if (threadIdx.x < st) red[threadIdx.x] += red[threadIdx.x + st];
    __syncthreads();
  }
  if (threadIdx.x == 0) { mx[row] = m; sm[row] = red[0]; }
}

// ---------------- kernel 5a: split-K partial KV = exp(K-max)^T @ V ----------------
// Grid: NSPLIT * (B*NHD) blocks, 9 waves each (3x3 16x16 tiles of the 48x48 product).
// kvp [NSPLIT, B*8, 48(e), 48(d)] f32 partials (deterministic reduction in 5b).
__global__ __launch_bounds__(288) void la_kv_partial(
    const float* __restrict__ kbuf, const _Float16* __restrict__ vbuf,
    const float* __restrict__ mx, float* __restrict__ kvp) {
  int split = blockIdx.x >> 6;               // 0..NSPLIT-1
  int bh = blockIdx.x & 63;
  int b = bh >> 3, h = bh & 7;
  int wave = threadIdx.x >> 5;               // 0..8
  int mt = wave / 3, nt = wave % 3;          // d-tile, e-tile
  int lane = threadIdx.x & 31;
  int r = lane & 15, hg = lane >> 4;
  int drow = h * NDH + mt * 16 + r;          // A row channel (d)
  int erow = h * NDH + nt * 16 + r;          // B^T row channel (e), lane n = r
  float mrow = mx[b * NC1 + drow];
  const float*    ka = kbuf + ((long long)b * NC1 + drow) * NL;
  const _Float16* vb = vbuf + ((long long)b * NC1 + erow) * NL;
  const int lchunk = NL / NSPLIT;            // 256 tokens per split
  v8f acc = {};
  for (int l = split * lchunk; l < (split + 1) * lchunk; l += 32) {
    const float* pa = ka + l + hg * 8;
    v4f f0 = *(const v4f*)(pa);
    v4f f1 = *(const v4f*)(pa + 4);
    v4f f2 = *(const v4f*)(pa + 16);
    v4f f3 = *(const v4f*)(pa + 20);
    v16h a;
    for (int i = 0; i < 4; i++) a[i]      = (_Float16)__expf(f0[i] - mrow);
    for (int i = 0; i < 4; i++) a[i + 4]  = (_Float16)__expf(f1[i] - mrow);
    for (int i = 0; i < 4; i++) a[i + 8]  = (_Float16)__expf(f2[i] - mrow);
    for (int i = 0; i < 4; i++) a[i + 12] = (_Float16)__expf(f3[i] - mrow);
    const _Float16* pb = vb + l + hg * 8;
    v16h bf = cat16(*(const v8h*)pb, *(const v8h*)(pb + 16));
    acc = wmma_f16(a, bf, acc);
  }
  // D: m = d (i + 8*hg within tile), n = e. Store kvp[e][d] contiguous in d.
  int e_abs = nt * 16 + (lane & 15);
  int d0 = mt * 16 + 8 * hg;
  float* dst = kvp + (((long long)split * 64 + bh) * NDH + e_abs) * NDH + d0;
  v4f o0, o1;
  for (int i = 0; i < 4; i++) o0[i] = acc[i];
  for (int i = 0; i < 4; i++) o1[i] = acc[i + 4];
  *(v4f*)dst = o0; *(v4f*)(dst + 4) = o1;
}

// ---------------- kernel 5b: reduce splits, scale by 1/sumexp, emit f16 kvT ----
__global__ __launch_bounds__(256) void la_kv_reduce(
    const float* __restrict__ kvp, const float* __restrict__ sm,
    _Float16* __restrict__ kvT) {
  const int per = NDH * NDH;                 // 2304
  const int total = 64 * per;                // 147456
  for (int i = blockIdx.x * 256 + threadIdx.x; i < total; i += gridDim.x * 256) {
    int bh = i / per;
    int rem = i - bh * per;
    int d = rem % NDH;
    float s = 0.f;
    for (int sp = 0; sp < NSPLIT; sp++)
      s += kvp[(long long)(sp * 64 + bh) * per + rem];
    int b = bh >> 3, h = bh & 7;
    kvT[i] = (_Float16)(s / sm[b * NC1 + h * NDH + d]);
  }
}

// ---------------- kernel 6: ctx = q@kv, out = ctx@Wo + bo + x1, LayerNorm ----
__global__ __launch_bounds__(256) void la_ctx_out_ln(
    const _Float16* __restrict__ qbuf, const _Float16* __restrict__ kvT,
    const _Float16* __restrict__ WoT, const float* __restrict__ bo,
    const float* __restrict__ x1, const float* __restrict__ gamma,
    const float* __restrict__ beta, float* __restrict__ out) {
  __shared__ __align__(16) _Float16 sctx[16 * NC1];
  __shared__ float sout[16 * NC1];
  __shared__ float sred[2][16][17];
  int b  = blockIdx.x / (NL / 16);
  int lt = blockIdx.x % (NL / 16);
  int wave = threadIdx.x >> 5;
  int lane = threadIdx.x & 31;
  int r = lane & 15, hg = lane >> 4;

  // ---- phase 1: ctx tiles (16 tokens x 16 e-channels), K = dh = 48 ----
  const _Float16* Q0 = qbuf + ((long long)b * NL + lt * 16) * NC1;
  for (int t = wave; t < 24; t += 8) {
    int h = t / 3, nt = t % 3;
    const _Float16* KV = kvT + ((long long)(b * NHD + h)) * NDH * NDH;
    v8f acc = {};
    acc = wmma_f16(load_a16(Q0 + h * NDH, NC1),
                   load_b16(KV + nt * 16 * NDH, NDH), acc);           // K 0..31
    acc = wmma_f16(load_a16_tail(Q0 + h * NDH + 32, NC1),
                   load_b16_tail(KV + nt * 16 * NDH + 32, NDH), acc); // K 32..47
    int ch = h * NDH + nt * 16 + (lane & 15);
    for (int i = 0; i < 8; i++) sctx[(i + 8 * hg) * NC1 + ch] = (_Float16)acc[i];
  }
  __syncthreads();

  // ---- phase 2: out projection + bias + residual ----
  const float* X1 = x1 + (long long)b * NC1 * NL + lt * 16;
  for (int t = wave; t < 24; t += 8) {
    v8f acc = {};
    for (int kk = 0; kk < NC1; kk += 32) {
      const _Float16* ap = &sctx[r * NC1 + kk + hg * 8];
      v16h a = cat16(*(const v8h*)ap, *(const v8h*)(ap + 16));
      v16h w = load_b16(WoT + (long long)t * 16 * NC1 + kk, NC1);
      acc = wmma_f16(a, w, acc);
    }
    int c = t * 16 + (lane & 15);
    float bias = bo[c];
    const float* xp = X1 + (long long)c * NL + 8 * hg;
    for (int i = 0; i < 8; i++)
      sout[(i + 8 * hg) * NC1 + c] = acc[i] + bias + xp[i];
  }
  __syncthreads();

  // ---- phase 3: LayerNorm over channels, store [B, C, L] ----
  int tok = threadIdx.x >> 4, idx = threadIdx.x & 15;
  float s = 0.f, s2 = 0.f;
  for (int c = idx; c < NC1; c += 16) {
    float v = sout[tok * NC1 + c];
    s += v; s2 += v * v;
  }
  sred[0][tok][idx] = s; sred[1][tok][idx] = s2;
  __syncthreads();
  if (idx == 0) {
    float a = 0.f, q2 = 0.f;
    for (int i = 0; i < 16; i++) { a += sred[0][tok][i]; q2 += sred[1][tok][i]; }
    sred[0][tok][16] = a; sred[1][tok][16] = q2;
  }
  __syncthreads();
  float mean = sred[0][tok][16] * (1.0f / NC1);
  float var  = sred[1][tok][16] * (1.0f / NC1) - mean * mean;
  float rstd = rsqrtf(var + 1e-5f);
  int l = lt * 16 + tok;
  for (int c = idx; c < NC1; c += 16) {
    float v = (sout[tok * NC1 + c] - mean) * rstd * gamma[c] + beta[c];
    out[((long long)b * NC1 + c) * NL + l] = v;
  }
}

extern "C" void kernel_launch(void* const* d_in, const int* in_sizes, int n_in,
                              void* d_out, int out_size, void* d_ws, size_t ws_size,
                              hipStream_t stream) {
  const float* x1    = (const float*)d_in[0];
  const float* x2    = (const float*)d_in[1];
  const float* Wq    = (const float*)d_in[2];
  const float* bq    = (const float*)d_in[3];
  const float* Wk    = (const float*)d_in[4];
  const float* bk    = (const float*)d_in[5];
  const float* Wv    = (const float*)d_in[6];
  const float* bv    = (const float*)d_in[7];
  const float* Wo    = (const float*)d_in[8];
  const float* bo    = (const float*)d_in[9];
  const float* gamma = (const float*)d_in[10];
  const float* beta  = (const float*)d_in[11];
  float* out = (float*)d_out;

  char* ws = (char*)d_ws;
  size_t off = 0;
  auto carve = [&](size_t bytes) {
    void* p = ws + off;
    off = (off + bytes + 255) & ~(size_t)255;
    return p;
  };
  _Float16* x1h  = (_Float16*)carve((size_t)NB * NL * NC1 * 2);
  _Float16* x2h  = (_Float16*)carve((size_t)NB * NL * NC2 * 2);
  _Float16* WqT  = (_Float16*)carve((size_t)NC1 * NC1 * 2);
  _Float16* WkT  = (_Float16*)carve((size_t)NC1 * NC2 * 2);
  _Float16* WvT  = (_Float16*)carve((size_t)NC1 * NC2 * 2);
  _Float16* WoT  = (_Float16*)carve((size_t)NC1 * NC1 * 2);
  _Float16* qbuf = (_Float16*)carve((size_t)NB * NL * NC1 * 2);
  float*    kbuf = (float*)   carve((size_t)NB * NC1 * NL * 4);
  _Float16* vbuf = (_Float16*)carve((size_t)NB * NC1 * NL * 2);
  float*    mxb  = (float*)   carve((size_t)NB * NC1 * 4);
  float*    smb  = (float*)   carve((size_t)NB * NC1 * 4);
  _Float16* kvT  = (_Float16*)carve((size_t)NB * NHD * NDH * NDH * 2);
  float*    kvp  = (float*)   carve((size_t)NSPLIT * NB * NHD * NDH * NDH * 4);

  float inv_scale = 1.0f / powf((float)NDH, 0.25f);

  // pack activations & weights
  la_cvt_tr<<<4096, 256, 0, stream>>>(x1, x1h, NC1, NL, NB);
  la_cvt_tr<<<4096, 256, 0, stream>>>(x2, x2h, NC2, NL, NB);
  la_cvt_tr<<<576, 256, 0, stream>>>(Wq, WqT, NC1, NC1, 1);
  la_cvt_tr<<<288, 256, 0, stream>>>(Wk, WkT, NC2, NC1, 1);
  la_cvt_tr<<<288, 256, 0, stream>>>(Wv, WvT, NC2, NC1, 1);
  la_cvt_tr<<<576, 256, 0, stream>>>(Wo, WoT, NC1, NC1, 1);

  // projections (4x N-blocked: 8 waves/block, 6 n-groups per token strip)
  la_proj_kv<<<NB * (NL / 16) * (NC1 / 64) / 8, 256, 0, stream>>>(
      x2h, WkT, WvT, bk, bv, kbuf, vbuf, inv_scale);
  la_proj_q<<<NB * (NL / 16) * (NC1 / 64) / 8, 256, 0, stream>>>(
      x1h, WqT, bq, qbuf, inv_scale);

  // softmax stats over token axis
  la_colstats<<<NB * NC1, 256, 0, stream>>>(kbuf, mxb, smb);

  // KV outer products: split-K partials + deterministic reduction
  la_kv_partial<<<NSPLIT * NB * NHD, 288, 0, stream>>>(kbuf, vbuf, mxb, kvp);
  la_kv_reduce<<<576, 256, 0, stream>>>(kvp, smb, kvT);

  // ctx, output projection, residual, LayerNorm, channel-major store
  la_ctx_out_ln<<<NB * (NL / 16), 256, 0, stream>>>(
      qbuf, kvT, WoT, bo, x1, gamma, beta, out);
}
